// LSTMCell_58016418234996
// MI455X (gfx1250) — matrix operand order repaired and checked by reading, not compiled, not run on hardware
//
#include <hip/hip_runtime.h>

// ---------------------------------------------------------------------------
// LSTM cell for MI455X (gfx1250, wave32, WMMA).
// gates = [x|h] @ [Wx;Wh] + b  via v_wmma_f32_16x16x32_bf16 (fp32 accum),
// fused sigmoid/tanh + state update, fully register-resident per 16x16 tile.
// Memory-bound target: ~544 MB streamed => ~23 us @ 23.3 TB/s.
// ---------------------------------------------------------------------------

typedef __attribute__((ext_vector_type(16))) __bf16        v16bf;
typedef __attribute__((ext_vector_type(8)))  float         v8f;
typedef __attribute__((ext_vector_type(4)))  unsigned int  u32x4;

#define B_ROWS      262144
#define IN_DIM      32
#define H_DIM       128
#define KDIM        160           // IN + H
#define NCOLS       512           // 4*H
#define ROWS_PER_WG 64            // 4 row-tiles of 16
#define A_STRIDE    168           // LDS bf16 row stride (84 dwords: conflict-lean)

static __device__ __forceinline__ unsigned short f2bf(float f) {
    union { float f; unsigned u; } v; v.f = f;
    unsigned r = v.u + 0x7FFFu + ((v.u >> 16) & 1u);   // round-to-nearest-even
    return (unsigned short)(r >> 16);
}
static __device__ __forceinline__ float sigm(float x) {
    return 1.0f / (1.0f + __expf(-x));
}
static __device__ __forceinline__ float tanh_fast(float x) {
    // tanh(x) = 1 - 2/(exp(2x)+1): monotone-safe for |x| large (no inf/inf)
    return 1.0f - 2.0f / (__expf(2.0f * x) + 1.0f);
}

// Pack W = [Wx;Wh] (row-major [160,512] fp32) into column-major bf16 Wt[512][160]
// so each lane's WMMA B-fragment is two contiguous 16-byte runs along K.
__global__ void prep_weights(const float* __restrict__ Wx,
                             const float* __restrict__ Wh,
                             unsigned short* __restrict__ Wt) {
    int j = blockIdx.x * blockDim.x + threadIdx.x;
    if (j >= NCOLS * KDIM) return;
    int n = j / KDIM;
    int k = j - n * KDIM;
    float v = (k < IN_DIM) ? Wx[k * NCOLS + n] : Wh[(k - IN_DIM) * NCOLS + n];
    Wt[j] = f2bf(v);
}

__global__ __launch_bounds__(256)
void lstm_kernel(const float* __restrict__ x, const float* __restrict__ h,
                 const float* __restrict__ c,
                 const unsigned short* __restrict__ Wt,
                 const float* __restrict__ bias,
                 float* __restrict__ h_out, float* __restrict__ c_out) {
    __shared__ __align__(16) unsigned short Ads[ROWS_PER_WG * A_STRIDE];

    const int tid  = threadIdx.x;
    const int lane = tid & 31;
    const int wave = tid >> 5;                 // 0..7: hidden slice [16w,16w+16)
    const int row0 = blockIdx.x * ROWS_PER_WG;

    // ---- Stage A panel (64 rows x 160 K) as bf16 into LDS, coalesced ----
    #pragma unroll
    for (int i = 0; i < 8; ++i) {              // x: 64x32 floats
        int e = tid + i * 256;
        int m = e >> 5, k = e & 31;
        Ads[m * A_STRIDE + k] = f2bf(x[(row0 + m) * IN_DIM + k]);
    }
    #pragma unroll
    for (int i = 0; i < 32; ++i) {             // h: 64x128 floats
        int e = tid + i * 256;
        int m = e >> 7, k = e & 127;
        Ads[m * A_STRIDE + IN_DIM + k] = f2bf(h[(row0 + m) * H_DIM + k]);
    }
    __syncthreads();

    const int nloc = lane & 15;                // output column within 16-tile
    const int hi   = (lane >> 4) & 1;          // half-wave select
    const int ncol = wave * 16 + nloc;         // hidden column 0..127

    // ---- Accumulators: acc[gate][row_tile], init to broadcast bias ----
    v8f acc[4][4];
    #pragma unroll
    for (int g = 0; g < 4; ++g) {
        float bv = bias[g * H_DIM + ncol];     // C/D layout: column fixed per lane
        v8f init = {bv, bv, bv, bv, bv, bv, bv, bv};
        #pragma unroll
        for (int rt = 0; rt < 4; ++rt) acc[g][rt] = init;
    }

    union Frag { u32x4 q[2]; v16bf v; };

    // ---- GEMM: K in 5 chunks of 32; B fragment reused across 4 row tiles ----
    #pragma unroll
    for (int kc = 0; kc < 5; ++kc) {
        // A fragments (16-bit A 16x32 layout): lane<16 -> K {0..7,16..23},
        // lane>=16 -> K {8..15,24..31} of this chunk, row M = lane%16.
        Frag af[4];
        const int ka = kc * 32 + hi * 8;
        #pragma unroll
        for (int rt = 0; rt < 4; ++rt) {
            const unsigned short* p = &Ads[(rt * 16 + nloc) * A_STRIDE + ka];
            af[rt].q[0] = *(const u32x4*)(p);
            af[rt].q[1] = *(const u32x4*)(p + 16);
        }
        const int kb = kc * 32 + hi * 8;       // B: lane<16 -> K {0..7,16..23}, else {8..15,24..31}
        #pragma unroll
        for (int g = 0; g < 4; ++g) {
            const unsigned short* pb = &Wt[(g * H_DIM + ncol) * KDIM + kb];
            Frag bf;
            bf.q[0] = *(const u32x4*)(pb);
            bf.q[1] = *(const u32x4*)(pb + 16);
            #pragma unroll
            for (int rt = 0; rt < 4; ++rt) {
                acc[g][rt] = __builtin_amdgcn_wmma_f32_16x16x32_bf16(
                    false, af[rt].v, false, bf.v, (short)0, acc[g][rt],
                    false, false);
            }
        }
    }

    // ---- Fused activations + state update, all in registers ----
    // C/D layout: VGPR r holds row M = r + 8*hi, column = ncol.
    #pragma unroll
    for (int rt = 0; rt < 4; ++rt) {
        #pragma unroll
        for (int vr = 0; vr < 8; ++vr) {
            int row = row0 + rt * 16 + hi * 8 + vr;
            float iv = sigm(acc[0][rt][vr]);
            float fv = sigm(acc[1][rt][vr]);
            float ov = sigm(acc[2][rt][vr]);
            float gv = tanh_fast(acc[3][rt][vr]);
            float cold = c[row * H_DIM + ncol];
            float cnew = fv * cold + iv * gv;
            float hnew = ov * tanh_fast(cnew);
            h_out[row * H_DIM + ncol] = hnew;
            c_out[row * H_DIM + ncol] = cnew;
        }
    }
}

extern "C" void kernel_launch(void* const* d_in, const int* in_sizes, int n_in,
                              void* d_out, int out_size, void* d_ws, size_t ws_size,
                              hipStream_t stream) {
    const float* x  = (const float*)d_in[0];
    const float* h  = (const float*)d_in[1];
    const float* c  = (const float*)d_in[2];
    const float* Wx = (const float*)d_in[3];
    const float* Wh = (const float*)d_in[4];
    const float* b  = (const float*)d_in[5];
    float* out = (float*)d_out;                       // [h_new | c_new] flat
    unsigned short* Wt = (unsigned short*)d_ws;       // 512*160 bf16 = 160 KB

    prep_weights<<<(NCOLS * KDIM + 255) / 256, 256, 0, stream>>>(Wx, Wh, Wt);
    lstm_kernel<<<B_ROWS / ROWS_PER_WG, 256, 0, stream>>>(
        x, h, c, Wt, b, out, out + (size_t)B_ROWS * H_DIM);
}
